// EGAD_19516331393172
// MI455X (gfx1250) — matrix operand order repaired and checked by reading, not compiled, not run on hardware
//
#include <hip/hip_runtime.h>
#include <hip/hip_bf16.h>
#include <math.h>
#include <stdint.h>

// Problem constants (match reference)
#define NNODES 4096
#define D1v 64
#define D2v 32
#define NHv 4
#define NCELLS 2

// GEMM tiling: 256 threads = 8 wave32s; block tile 128x128, K step 32.
#define BM 128
#define BN 128
#define KT 32
#define KTP 40   // padded LDS row stride (shorts): 80B = 5*16B -> aligned + bank spread

// Optional CDNA5 async global->LDS path (ASYNCcnt-tracked), with safe fallback.
#if defined(__AMDGCN__) && __has_builtin(__builtin_amdgcn_global_load_async_to_lds_b128)
#define USE_ASYNC_LDS 1
#else
#define USE_ASYNC_LDS 0
#endif

typedef __attribute__((ext_vector_type(16))) __bf16 v16bf;
typedef __attribute__((ext_vector_type(8)))  float  v8f;
typedef __attribute__((ext_vector_type(4)))  int    v4i_t;
// AS-qualified pointee types for the async builtin (param0 is v4i in AS(1))
typedef __attribute__((address_space(1))) v4i_t gv4i;
typedef __attribute__((address_space(3))) v4i_t lv4i;

union Frag { unsigned int u[8]; v16bf v; };

__device__ __forceinline__ unsigned short f2bf(float x) {
  unsigned int u = __float_as_uint(x);
  u += 0x7FFFu + ((u >> 16) & 1u);           // round-to-nearest-even
  return (unsigned short)(u >> 16);
}
__device__ __forceinline__ float eluf(float x)  { return x > 0.0f ? x : expm1f(x); }
__device__ __forceinline__ float lrelu(float x) { return x > 0.0f ? x : 0.2f * x; }

// ---------------- elementwise / small kernels ----------------

__global__ __launch_bounds__(256) void zero_bf16(unsigned short* dst, long n) {
  long gid = (long)blockIdx.x * blockDim.x + threadIdx.x;
  if (gid < n) dst[gid] = 0;
}

__global__ __launch_bounds__(256) void cvt_bf16(const float* __restrict__ src,
                                                unsigned short* __restrict__ dst, long n) {
  long stride = (long)gridDim.x * blockDim.x;
  for (long i = (long)blockIdx.x * blockDim.x + threadIdx.x; i < n; i += stride)
    dst[i] = f2bf(src[i]);
}

// C[M,Nc] = A[M,K] @ B[K,Nc] in fp32 (tiny projections), plus bf16 copy into a
// 128-wide zero-padded buffer feeding the WMMA GEMM.
__global__ __launch_bounds__(256) void small_nn(const float* __restrict__ A,
                                                const float* __restrict__ B,
                                                float* __restrict__ Cf,
                                                unsigned short* __restrict__ Cbfp,
                                                int M, int K, int Nc) {
  long gid = (long)blockIdx.x * blockDim.x + threadIdx.x;
  if (gid >= (long)M * Nc) return;
  int row = (int)(gid / Nc), col = (int)(gid % Nc);
  float acc = 0.0f;
  for (int k = 0; k < K; ++k)
    acc += A[(size_t)row * K + k] * B[(size_t)k * Nc + col];
  Cf[(size_t)row * Nc + col] = acc;
  if (Cbfp) Cbfp[(size_t)row * 128 + col] = f2bf(acc);
}

// f1[r] = h[r,:] @ a[:64]; f2[r] = h[r,:] @ a[64:]
__global__ __launch_bounds__(256) void compute_f(const float* __restrict__ h,
                                                 const float* __restrict__ a,
                                                 float* __restrict__ f1,
                                                 float* __restrict__ f2) {
  int row = blockIdx.x * blockDim.x + threadIdx.x;
  if (row >= NNODES) return;
  float s1 = 0.0f, s2 = 0.0f;
  for (int d = 0; d < D1v; ++d) {
    float v = h[(size_t)row * D1v + d];
    s1 += v * a[d];
    s2 += v * a[D1v + d];
  }
  f1[row] = s1; f2[row] = s2;
}

// e[i,j] = leaky_relu(f1[i] + f2[j]) stored bf16 (B operand of adj @ e)
__global__ __launch_bounds__(256) void build_e(const float* __restrict__ f1,
                                               const float* __restrict__ f2,
                                               unsigned short* __restrict__ e) {
  long gid = (long)blockIdx.x * blockDim.x + threadIdx.x;
  if (gid >= (long)NNODES * NNODES) return;
  int i = (int)(gid / NNODES), j = (int)(gid % NNODES);
  e[gid] = f2bf(lrelu(f1[i] + f2[j]));
}

// row softmax of S (fp32) -> att (bf16), one block per row
__global__ __launch_bounds__(256) void row_softmax_bf16(const float* __restrict__ S,
                                                        unsigned short* __restrict__ att) {
  __shared__ float red[256];
  const int row = blockIdx.x, tid = threadIdx.x;
  const float* s = S + (size_t)row * NNODES;
  float mx = -INFINITY;
  for (int j = tid; j < NNODES; j += 256) mx = fmaxf(mx, s[j]);
  red[tid] = mx; __syncthreads();
  for (int st = 128; st > 0; st >>= 1) {
    if (tid < st) red[tid] = fmaxf(red[tid], red[tid + st]);
    __syncthreads();
  }
  mx = red[0]; __syncthreads();
  float sum = 0.0f;
  for (int j = tid; j < NNODES; j += 256) sum += __expf(s[j] - mx);
  red[tid] = sum; __syncthreads();
  for (int st = 128; st > 0; st >>= 1) {
    if (tid < st) red[tid] += red[tid + st];
    __syncthreads();
  }
  float inv = 1.0f / red[0];
  for (int j = tid; j < NNODES; j += 256)
    att[(size_t)row * NNODES + j] = f2bf(__expf(s[j] - mx) * inv);
}

__global__ __launch_bounds__(256) void elu_concat(const float* __restrict__ hp128,
                                                  float* __restrict__ cat, int head) {
  long gid = (long)blockIdx.x * blockDim.x + threadIdx.x;
  if (gid >= (long)NNODES * D1v) return;
  int row = (int)(gid / D1v), col = (int)(gid % D1v);
  cat[(size_t)row * (NHv * D1v) + head * D1v + col] = eluf(hp128[(size_t)row * 128 + col]);
}

__global__ __launch_bounds__(256) void elu_store(const float* __restrict__ hp128,
                                                 float* __restrict__ x1,
                                                 unsigned short* __restrict__ x1bp) {
  long gid = (long)blockIdx.x * blockDim.x + threadIdx.x;
  if (gid >= (long)NNODES * D1v) return;
  int row = (int)(gid / D1v), col = (int)(gid % D1v);
  float v = eluf(hp128[(size_t)row * 128 + col]);
  x1[(size_t)row * D1v + col] = v;
  x1bp[(size_t)row * 128 + col] = f2bf(v);
}

__global__ __launch_bounds__(256) void relu_cols(const float* __restrict__ hp128,
                                                 float* __restrict__ h1) {
  long gid = (long)blockIdx.x * blockDim.x + threadIdx.x;
  if (gid >= (long)NNODES * D1v) return;
  int row = (int)(gid / D1v), col = (int)(gid % D1v);
  h1[(size_t)row * D1v + col] = fmaxf(hp128[(size_t)row * 128 + col], 0.0f);
}

__global__ __launch_bounds__(256) void copy_out(const float* __restrict__ hp128,
                                                float* __restrict__ out) {
  long gid = (long)blockIdx.x * blockDim.x + threadIdx.x;
  if (gid >= (long)NNODES * D2v) return;
  int row = (int)(gid / D2v), col = (int)(gid % D2v);
  out[gid] = hp128[(size_t)row * 128 + col];
}

// ---------------- WMMA GEMM: C(f32) = A(bf16) @ B(bf16) ----------------
// Row-major, lda=K, ldb=ldc=N. M%128==0, N%128==0, K%32==0.
// 8 wave32s per block; each wave owns a 32x64 C region = 2x4 16x16 WMMA tiles.
// A tile staged [row][k] (async global->LDS when available); B tile staged
// TRANSPOSED [col][k] so both A and B fragments are two contiguous b128 LDS
// reads per the ISA 7.12.2 bf16 lane layouts.
__global__ __launch_bounds__(256) void gemm_bf16_wmma(const unsigned short* __restrict__ A,
                                                      const unsigned short* __restrict__ B,
                                                      float* __restrict__ C,
                                                      int M, int N, int K) {
  __shared__ __align__(16) unsigned short As[BM * KTP];   // 10 KB
  __shared__ __align__(16) unsigned short Bs[BN * KTP];   // 10 KB, [col][k]
  const int tid  = threadIdx.x;
  const int lane = tid & 31;
  const int wave = tid >> 5;
  const int wm   = wave & 3;        // 0..3: 32-row group
  const int wn   = wave >> 2;       // 0..1: 64-col group
  const int m16  = lane & 15;
  const int half = lane >> 4;       // K-half select per ISA bf16 lane layout
  const int blockRow = blockIdx.y * BM;
  const int blockCol = blockIdx.x * BN;

  v8f acc[2][4] = {};

  for (int k0 = 0; k0 < K; k0 += KT) {
    // ---- stage A tile 128x32 (4 x b128 per row) ----
#pragma unroll
    for (int it = 0; it < 2; ++it) {
      int i = tid + it * 256;
      int r = i >> 2, seg = i & 3;
      const unsigned short* gsrc = &A[(size_t)(blockRow + r) * K + k0 + seg * 8];
      unsigned short* ldst = &As[r * KTP + seg * 8];
#if USE_ASYNC_LDS
      __builtin_amdgcn_global_load_async_to_lds_b128(
          (gv4i*)(uintptr_t)gsrc,
          (lv4i*)(unsigned int)(uintptr_t)ldst,
          0, 0);
#else
      *(uint4*)ldst = *(const uint4*)gsrc;
#endif
    }
    // ---- stage B tile 32x128, transposed into Bs[col][k] ----
#pragma unroll
    for (int it = 0; it < 2; ++it) {
      int i = tid + it * 256;
      int r = i >> 4, seg = i & 15;
      uint4 d = *(const uint4*)&B[(size_t)(k0 + r) * N + blockCol + seg * 8];
      const unsigned short* s = (const unsigned short*)&d;
#pragma unroll
      for (int j = 0; j < 8; ++j)
        Bs[(seg * 8 + j) * KTP + r] = s[j];
    }
    if (k0 + KT < K) {  // speculative prefetch of next K slab -> global_prefetch
      __builtin_prefetch(&B[(size_t)(k0 + KT + (tid >> 4)) * N + blockCol + (tid & 15) * 8], 0, 0);
#if !USE_ASYNC_LDS
      __builtin_prefetch(&A[(size_t)(blockRow + (tid >> 2)) * K + k0 + KT + (tid & 3) * 8], 0, 0);
#endif
    }
#if USE_ASYNC_LDS
#if __has_builtin(__builtin_amdgcn_s_wait_asynccnt)
    __builtin_amdgcn_s_wait_asynccnt(0);
#else
    asm volatile("s_wait_asynccnt 0" ::: "memory");
#endif
#endif
    __syncthreads();

    // ---- fragments: 2 x b128 LDS reads each ----
    // A (16x32): lanes 0-15 K 0-7 (v0-3) / 16-23 (v4-7); lanes 16-31 K 8-15 / 24-31.
    Frag af[2], bf[4];
#pragma unroll
    for (int tm = 0; tm < 2; ++tm) {
      const unsigned short* pa = &As[(wm * 32 + tm * 16 + m16) * KTP + half * 8];
      uint4 lo = *(const uint4*)pa;
      uint4 hi = *(const uint4*)(pa + 16);
      af[tm].u[0] = lo.x; af[tm].u[1] = lo.y; af[tm].u[2] = lo.z; af[tm].u[3] = lo.w;
      af[tm].u[4] = hi.x; af[tm].u[5] = hi.y; af[tm].u[6] = hi.z; af[tm].u[7] = hi.w;
    }
    // B (32x16): lanes 0-15 K 0-7 (v0-3) / 16-23 (v4-7); lanes 16-31 K 8-15 / 24-31.
#pragma unroll
    for (int tn = 0; tn < 4; ++tn) {
      int col = wn * 64 + tn * 16 + m16;
      const unsigned short* pb = &Bs[col * KTP + half * 8];
      uint4 lo = *(const uint4*)pb;
      uint4 hi = *(const uint4*)(pb + 16);
      bf[tn].u[0] = lo.x; bf[tn].u[1] = lo.y; bf[tn].u[2] = lo.z; bf[tn].u[3] = lo.w;
      bf[tn].u[4] = hi.x; bf[tn].u[5] = hi.y; bf[tn].u[6] = hi.z; bf[tn].u[7] = hi.w;
    }
#pragma unroll
    for (int tm = 0; tm < 2; ++tm)
#pragma unroll
      for (int tn = 0; tn < 4; ++tn)
        acc[tm][tn] = __builtin_amdgcn_wmma_f32_16x16x32_bf16(
            false, af[tm].v, false, bf[tn].v, (short)0, acc[tm][tn], false, false);
    __syncthreads();
  }

  // D layout: lanes 0-15 -> N=lane, M=r; lanes 16-31 -> N=lane-16, M=r+8
#pragma unroll
  for (int tm = 0; tm < 2; ++tm) {
#pragma unroll
    for (int tn = 0; tn < 4; ++tn) {
      int baseRow = blockRow + wm * 32 + tm * 16 + half * 8;
      int col     = blockCol + wn * 64 + tn * 16 + m16;
#pragma unroll
      for (int r = 0; r < 8; ++r)
        C[(size_t)(baseRow + r) * N + col] = acc[tm][tn][r];
    }
  }
}

// ---------------- host orchestration ----------------

extern "C" void kernel_launch(void* const* d_in, const int* in_sizes, int n_in,
                              void* d_out, int out_size, void* d_ws, size_t ws_size,
                              hipStream_t stream) {
  (void)in_sizes; (void)n_in; (void)out_size; (void)ws_size;
  const float* x      = (const float*)d_in[0];
  const float* adj    = (const float*)d_in[1];
  const float* w_init = (const float*)d_in[2];
  const float* head_H = (const float*)d_in[3];
  const float* head_a = (const float*)d_in[4];
  const float* out_H  = (const float*)d_in[5];
  const float* out_a  = (const float*)d_in[6];
  const float* l2_w   = (const float*)d_in[7];
  float* out = (float*)d_out;

  // Workspace arena (requires ~185 MB; all buffers 256B aligned)
  char* p = (char*)d_ws;
  auto alloc = [&](size_t bytes) { char* r = p; p += (bytes + 255) & ~(size_t)255; return r; };
  const size_t NNsz = (size_t)NNODES * NNODES;
  unsigned short* adjB = (unsigned short*)alloc(NNsz * 2);           // adj, bf16
  unsigned short* EB   = (unsigned short*)alloc(NNsz * 2);           // e, then att
  float*          S    = (float*)alloc(NNsz * 4);                    // adj @ e
  unsigned short* xB   = (unsigned short*)alloc(NNsz * 2);           // x[i], bf16
  float*          hF   = (float*)alloc((size_t)NNODES * D1v * 4);
  unsigned short* hBp  = (unsigned short*)alloc((size_t)NNODES * 128 * 2);
  float*          f1   = (float*)alloc((size_t)NNODES * 4);
  float*          f2   = (float*)alloc((size_t)NNODES * 4);
  float*          HPp  = (float*)alloc((size_t)NNODES * 128 * 4);
  float*          cat  = (float*)alloc((size_t)NNODES * NHv * D1v * 4);
  float*          x1F  = (float*)alloc((size_t)NNODES * D1v * 4);
  unsigned short* x1Bp = (unsigned short*)alloc((size_t)NNODES * 128 * 2);
  float*          TF   = (float*)alloc((size_t)NNODES * 128 * 4);
  unsigned short* tB   = (unsigned short*)alloc((size_t)NNODES * 128 * 2);
  float*          h1F  = (float*)alloc((size_t)NNODES * D1v * 4);
  float*          uF   = (float*)alloc((size_t)NNODES * D2v * 4);
  unsigned short* uBp  = (unsigned short*)alloc((size_t)NNODES * 128 * 2);

  const int T = 256;
  const long pad128 = (long)NNODES * 128;
  // zero padded bf16 buffers so pad columns contribute exact zeros in WMMA
  zero_bf16<<<(int)((pad128 + T - 1) / T), T, 0, stream>>>(hBp, pad128);
  zero_bf16<<<(int)((pad128 + T - 1) / T), T, 0, stream>>>(x1Bp, pad128);
  zero_bf16<<<(int)((pad128 + T - 1) / T), T, 0, stream>>>(uBp, pad128);

  const dim3 gBig(NNODES / BN, NNODES / BM);   // 32x32 blocks
  const dim3 gSkin(1, NNODES / BM);            // N=128 GEMMs

  const float* prev = w_init;
  for (int c = 0; c < NCELLS; ++c) {
    const float* adj_c = adj + (size_t)c * NNsz;
    cvt_bf16<<<4096, T, 0, stream>>>(adj_c, adjB, (long)NNsz);

    for (int hd = 0; hd < NHv; ++hd) {
      const float* Hm = head_H + ((size_t)c * NHv + hd) * D1v * D1v;
      const float* av = head_a + ((size_t)c * NHv + hd) * 2 * D1v;
      small_nn<<<NNODES * D1v / T, T, 0, stream>>>(prev, Hm, hF, hBp, NNODES, D1v, D1v);
      compute_f<<<NNODES / T, T, 0, stream>>>(hF, av, f1, f2);
      build_e<<<(int)(NNsz / T), T, 0, stream>>>(f1, f2, EB);
      gemm_bf16_wmma<<<gBig, T, 0, stream>>>(adjB, EB, S, NNODES, NNODES, NNODES);
      row_softmax_bf16<<<NNODES, T, 0, stream>>>(S, EB);          // att -> EB
      gemm_bf16_wmma<<<gSkin, T, 0, stream>>>(EB, hBp, HPp, NNODES, 128, NNODES);
      elu_concat<<<NNODES * D1v / T, T, 0, stream>>>(HPp, cat, hd);
    }
    // out-attention GAT (concat=False), then x1 = elu(hp)
    small_nn<<<NNODES * D1v / T, T, 0, stream>>>(cat, out_H + (size_t)c * NHv * D1v * D1v,
                                                 hF, hBp, NNODES, NHv * D1v, D1v);
    compute_f<<<NNODES / T, T, 0, stream>>>(hF, out_a + (size_t)c * 2 * D1v, f1, f2);
    build_e<<<(int)(NNsz / T), T, 0, stream>>>(f1, f2, EB);
    gemm_bf16_wmma<<<gBig, T, 0, stream>>>(adjB, EB, S, NNODES, NNODES, NNODES);
    row_softmax_bf16<<<NNODES, T, 0, stream>>>(S, EB);
    gemm_bf16_wmma<<<gSkin, T, 0, stream>>>(EB, hBp, HPp, NNODES, 128, NNODES);
    elu_store<<<NNODES * D1v / T, T, 0, stream>>>(HPp, x1F, x1Bp);

    if (c == NCELLS - 1) {   // only the final cell's h2 is returned
      cvt_bf16<<<4096, T, 0, stream>>>(x + (size_t)c * NNsz, xB, (long)NNsz);
      gemm_bf16_wmma<<<gSkin, T, 0, stream>>>(xB, x1Bp, TF, NNODES, 128, NNODES);
      cvt_bf16<<<2048, T, 0, stream>>>(TF, tB, pad128);
      gemm_bf16_wmma<<<gSkin, T, 0, stream>>>(adjB, tB, HPp, NNODES, 128, NNODES);
      relu_cols<<<NNODES * D1v / T, T, 0, stream>>>(HPp, h1F);
      small_nn<<<NNODES * D2v / T, T, 0, stream>>>(h1F, l2_w + (size_t)c * D1v * D2v,
                                                   uF, uBp, NNODES, D1v, D2v);
      gemm_bf16_wmma<<<gSkin, T, 0, stream>>>(adjB, uBp, HPp, NNODES, 128, NNODES);
      copy_out<<<NNODES * D2v / T, T, 0, stream>>>(HPp, out);
    }
    prev = x1F;
  }
}
